// SUPRARecurrentGated_58102317580756
// MI455X (gfx1250) — matrix operand order repaired and checked by reading, not compiled, not run on hardware
//
#include <hip/hip_runtime.h>

typedef __attribute__((ext_vector_type(16))) _Float16 v16h;
typedef __attribute__((ext_vector_type(8)))  float    v8f;
typedef __attribute__((ext_vector_type(4)))  unsigned int u32x4;
typedef __attribute__((ext_vector_type(8)))  int      i32x8;
typedef __attribute__((ext_vector_type(4)))  int      i32x4;

#define B_   8
#define S_   4096
#define D_   512
#define T_   32            // chunk length (== WMMA K)
#define NC_  (S_ / T_)     // 128 chunks
#define DJ_  64            // state columns per workgroup
#define TI_  128           // i-tile size within a chunk
#define NIT_ (D_ / TI_)    // 4 i-tiles
#define EPS_ 1e-6f

// ---------------- TDM: async 2D f32 tile load, global -> LDS ---------------
// Builds a Tensor DMA Descriptor (D#) per CDNA5 ISA and
// issues TENSOR_LOAD_TO_LDS. 2D tile: tile_h rows x tile_w f32, row stride
// row_stride_elems. Tracked by TENSORcnt; wait with s_wait_tensorcnt.
__device__ __forceinline__ void tdm_load_2d_f32(const float* gsrc, void* ldst,
                                                int tile_w, int tile_h,
                                                int row_stride_elems) {
  const unsigned long long ga = (unsigned long long)(size_t)gsrc;
  const unsigned lds = (unsigned)(size_t)ldst;
  u32x4 g0;
  g0[0] = 1u;                                          // count=1 (valid), user desc
  g0[1] = lds;                                         // lds_addr
  g0[2] = (unsigned)(ga & 0xffffffffull);              // global_addr[31:0]
  g0[3] = (unsigned)((ga >> 32) & 0x1ffffffull)        // global_addr[56:32]
        | (2u << 30);                                  // type=2 ("image")
  i32x8 g1;
  g1[0] = (int)(2u << 16);                             // data_size=2 -> 4 bytes
  g1[1] = (int)((unsigned)tile_w << 16);               // tensor_dim0[15:0]
  g1[2] = (int)((((unsigned)tile_w >> 16) & 0xffffu)   // tensor_dim0[31:16]
        | ((unsigned)tile_h << 16));                   // tensor_dim1[15:0]
  g1[3] = (int)((((unsigned)tile_h >> 16) & 0xffffu)   // tensor_dim1[31:16]
        | ((unsigned)tile_w << 16));                   // tile_dim0
  g1[4] = (int)((unsigned)tile_h & 0xffffu);           // tile_dim1 (tile_dim2=0)
  g1[5] = (int)(unsigned)row_stride_elems;             // tensor_dim0_stride[31:0]
  g1[6] = 0;                                           // stride[47:32], dim1_stride lo
  g1[7] = 0;
  i32x4 z4 = {0, 0, 0, 0};
#if __clang_major__ >= 23
  i32x8 z8 = {0, 0, 0, 0, 0, 0, 0, 0};
  __builtin_amdgcn_tensor_load_to_lds(g0, g1, z4, z4, z8, 0);
#else
  __builtin_amdgcn_tensor_load_to_lds(g0, g1, z4, z4, 0);
#endif
}

// ---------------- WMMA fragment loaders (f32 source -> f16 fragment) -------
// A fragment, logical A[m][k] = src[m*ld + k], 16x32.
__device__ __forceinline__ v16h load_a_frag_f32(const float* src, int ld, int lane) {
  const int m    = lane & 15;
  const int koff = (lane >> 4) << 4;   // 0 or 16
  const float* row = src + (size_t)m * ld;
  v16h a;
#pragma unroll
  for (int v = 0; v < 4; ++v) {
    int k  = koff + 2 * v;
    a[2*v]     = (_Float16)row[k];
    a[2*v + 1] = (_Float16)row[k + 1];
    int k2 = 8 + koff + 2 * v;
    a[8 + 2*v]     = (_Float16)row[k2];
    a[8 + 2*v + 1] = (_Float16)row[k2 + 1];
  }
  return a;
}

// B fragment, logical B[k][n] = src[k*ld + n], 32x16.
__device__ __forceinline__ v16h load_b_frag_f32(const float* src, int ld, int lane) {
  const int n  = lane & 15;
  const int kb = (lane >> 4) << 4;     // 0 or 16
  v16h b;
#pragma unroll
  for (int v = 0; v < 8; ++v) {
    int k = kb + 2 * v;
    b[2*v]     = (_Float16)src[(size_t)k * ld + n];
    b[2*v + 1] = (_Float16)src[(size_t)(k + 1) * ld + n];
  }
  return b;
}

// B fragment where logical B[k][n] = src[n*ld + k]  (B = src^T), 32x16.
__device__ __forceinline__ v16h load_bT_frag_f32(const float* src, int ld, int lane) {
  const int n  = lane & 15;
  const int kb = (lane >> 4) << 4;
  const float* col = src + (size_t)n * ld;
  v16h b;
#pragma unroll
  for (int v = 0; v < 8; ++v) {
    int k = kb + 2 * v;
    b[2*v]     = (_Float16)col[k];
    b[2*v + 1] = (_Float16)col[k + 1];
  }
  return b;
}

// ---------------- Kernel 1: fused QKVA projections (WMMA GEMM) -------------
__global__ __launch_bounds__(256) void suprag_proj_kernel(
    const float* __restrict__ x,
    const float* __restrict__ Wq, const float* __restrict__ bq,
    const float* __restrict__ Wk, const float* __restrict__ bk,
    const float* __restrict__ Wv, const float* __restrict__ bv,
    const float* __restrict__ Wa, const float* __restrict__ ba,
    float* __restrict__ qo, float* __restrict__ ko,
    float* __restrict__ vo, float* __restrict__ ao)
{
  const int lane = threadIdx.x & 31;
  const int wave = threadIdx.x >> 5;
  const int m0   = blockIdx.x * 16;
  const int eg   = blockIdx.y * 128 + wave * 16;
  const int proj = eg >> 9;                       // 0:q 1:k 2:v 3:a
  const int e0   = eg & (D_ - 1);

  const float* W    = (proj == 0) ? Wq : (proj == 1) ? Wk : (proj == 2) ? Wv : Wa;
  const float* bias = (proj == 0) ? bq : (proj == 1) ? bk : (proj == 2) ? bv : ba;
  float*       out  = (proj == 0) ? qo : (proj == 1) ? ko : (proj == 2) ? vo : ao;

  v8f acc = {};
#pragma unroll 4
  for (int k0 = 0; k0 < D_; k0 += 32) {
    v16h af = load_a_frag_f32(x + (size_t)m0 * D_ + k0, D_, lane);
    v16h bf = load_bT_frag_f32(W + (size_t)e0 * D_ + k0, D_, lane);
    acc = __builtin_amdgcn_wmma_f32_16x16x32_f16(false, af, false, bf,
                                                 (short)0, acc, false, false);
  }

  const int n     = lane & 15;
  const int rbase = (lane >> 4) << 3;
  const float bv_ = bias[e0 + n];
#pragma unroll
  for (int r = 0; r < 8; ++r) {
    const int m = m0 + r + rbase;
    float val = acc[r] + bv_;
    if (proj <= 1)      val = fmaxf(val, 0.0f);              // relu q,k
    else if (proj == 3) val = 1.0f / (1.0f + __expf(-val));  // sigmoid a
    out[(size_t)m * D_ + (e0 + n)] = val;
  }
}

// ---------------- Kernel 2: chunked scan, TDM double-buffered --------------
__global__ __launch_bounds__(256) void suprag_scan_kernel(
    const float* __restrict__ qb, const float* __restrict__ kbf,
    const float* __restrict__ vb, const float* __restrict__ ab,
    float* __restrict__ out)
{
  extern __shared__ char smem_raw[];
  float* Ssm = (float*)smem_raw;            // [D_][DJ_] state (f32)
  float* Vb0 = Ssm + D_ * DJ_;              // [T_][DJ_] v chunk, buffer 0
  float* Vb1 = Vb0 + T_ * DJ_;              // buffer 1
  float* Qt0 = Vb1 + T_ * DJ_;              // [T_][TI_] q/k/a tiles, buffer 0
  float* Kt0 = Qt0 + T_ * TI_;
  float* At0 = Kt0 + T_ * TI_;
  float* Qt1 = At0 + T_ * TI_;              // buffer 1
  float* Kt1 = Qt1 + T_ * TI_;
  float* At1 = Kt1 + T_ * TI_;
  float* Gsm   = At1 + T_ * TI_;            // forward cumprod of a
  float* QGsm  = Gsm  + T_ * TI_;           // q * G_t
  float* KDsm  = QGsm + T_ * TI_;           // k / G_t (clamped)
  float* KBsm  = KDsm + T_ * TI_;           // k * (G_T/G_t)
  float* Zsm   = KBsm + T_ * TI_;           // [D_] z state
  float* Psm   = Zsm + D_;                  // [T_][T_] intra-chunk attn
  float* DENsm = Psm + T_ * T_;             // [T_]

  float* Vbuf[2] = {Vb0, Vb1};
  float* Qbuf[2] = {Qt0, Qt1};
  float* Kbuf[2] = {Kt0, Kt1};
  float* Abuf[2] = {At0, At1};

  const int tid  = threadIdx.x;
  const int lane = tid & 31;
  const int wave = tid >> 5;
  const int jt   = blockIdx.x;              // 0..7
  const int b    = blockIdx.y;              // 0..7
  const int j0   = jt * DJ_;
  const size_t base = (size_t)b * S_ * D_;
  const bool issuer = (wave == 0);

  for (int idx = tid; idx < D_ * DJ_; idx += 256) Ssm[idx] = 0.0f;
  for (int idx = tid; idx < D_;       idx += 256) Zsm[idx] = 0.0f;
  __syncthreads();

  // Prime the pipeline: V of chunk 0 + q/k/a tile (c=0, it=0).
  if (issuer) {
    tdm_load_2d_f32(vb + base + j0, Vbuf[0], DJ_, T_, D_);
    const size_t g0off = base;              // t0=0, i0=0
    tdm_load_2d_f32(qb  + g0off, Qbuf[0], TI_, T_, D_);
    tdm_load_2d_f32(kbf + g0off, Kbuf[0], TI_, T_, D_);
    tdm_load_2d_f32(ab  + g0off, Abuf[0], TI_, T_, D_);
  }

  const int mt = wave >> 2;                 // 0..1 (output t-tile)
  const int nt = wave & 3;                  // 0..3 (output j-tile)

  for (int c = 0; c < NC_; ++c) {
    const int t0 = c * T_;
    float* Vcur = Vbuf[c & 1];

    if (tid < T_) DENsm[tid] = 0.0f;

    v8f  nacc = {};
    float pacc[4] = {0.f, 0.f, 0.f, 0.f};

    for (int it = 0; it < NIT_; ++it) {
      const int i0 = it * TI_;
      const int p  = it & 1;                // current tile buffer parity

      // DMA of current stage complete, publish to all waves.
      if (issuer) __builtin_amdgcn_s_wait_tensorcnt(0);
      __syncthreads();

      // Kick off next stage's DMA; overlaps with all compute below.
      if (issuer) {
        if (it + 1 < NIT_) {
          const size_t g = base + (size_t)t0 * D_ + (i0 + TI_);
          tdm_load_2d_f32(qb  + g, Qbuf[p ^ 1], TI_, T_, D_);
          tdm_load_2d_f32(kbf + g, Kbuf[p ^ 1], TI_, T_, D_);
          tdm_load_2d_f32(ab  + g, Abuf[p ^ 1], TI_, T_, D_);
        } else if (c + 1 < NC_) {
          const size_t g = base + (size_t)(t0 + T_) * D_;
          tdm_load_2d_f32(vb + g + j0, Vbuf[(c + 1) & 1], DJ_, T_, D_);
          tdm_load_2d_f32(qb  + g, Qbuf[p ^ 1], TI_, T_, D_);
          tdm_load_2d_f32(kbf + g, Kbuf[p ^ 1], TI_, T_, D_);
          tdm_load_2d_f32(ab  + g, Abuf[p ^ 1], TI_, T_, D_);
        }
      }

      const float* Q = Qbuf[p];
      const float* K = Kbuf[p];
      const float* A = Abuf[p];

      // Per-feature gate cumprods (division-free except KD).
      if (tid < TI_) {
        const int i = tid;
        float g = 1.0f;
#pragma unroll 1
        for (int t = 0; t < T_; ++t) {
          g *= A[t * TI_ + i];
          Gsm[t * TI_ + i]  = g;
          QGsm[t * TI_ + i] = Q[t * TI_ + i] * g;
          KDsm[t * TI_ + i] = K[t * TI_ + i] / fmaxf(g, 1e-30f);
        }
        float w = 1.0f;
#pragma unroll 1
        for (int t = T_ - 1; t >= 0; --t) {
          KBsm[t * TI_ + i] = K[t * TI_ + i] * w;
          w *= A[t * TI_ + i];
        }
      }
      __syncthreads();

      // z recurrence + denominator partials (wave reduce -> one atomic).
      if (tid < TI_) {
        const int i = tid;
        float z = Zsm[i0 + i];
#pragma unroll 1
        for (int t = 0; t < T_; ++t) {
          z = A[t * TI_ + i] * z + K[t * TI_ + i];
          float d = Q[t * TI_ + i] * z;
#pragma unroll
          for (int off = 16; off > 0; off >>= 1) d += __shfl_xor(d, off, 32);
          if (lane == 0) atomicAdd(&DENsm[t], d);
        }
        Zsm[i0 + i] = z;
      }

      // intra-chunk P partials: 1024 (t,tau) pairs, 4 per thread.
#pragma unroll
      for (int pp = 0; pp < 4; ++pp) {
        const int pr = tid + 256 * pp;
        const int t = pr >> 5, tau = pr & 31;
        if (tau <= t) {
          const float* qr = QGsm + t * TI_;
          const float* kr = KDsm + tau * TI_;
          float s = 0.0f;
#pragma unroll 4
          for (int i = 0; i < TI_; ++i) s += qr[i] * kr[i];
          pacc[pp] += s;
        }
      }

      // Phase A: num_inter += q~ (32xK) @ S0 (Kx64) for this K slice.
#pragma unroll
      for (int ks = 0; ks < TI_ / 32; ++ks) {
        const float* asrc = QGsm + (size_t)(mt * 16) * TI_ + ks * 32;
        v16h af = load_a_frag_f32(asrc, TI_, lane);
        const float* bsrc = Ssm + (size_t)(i0 + ks * 32) * DJ_ + nt * 16;
        v16h bf = load_b_frag_f32(bsrc, DJ_, lane);
        nacc = __builtin_amdgcn_wmma_f32_16x16x32_f16(false, af, false, bf,
                                                      (short)0, nacc, false, false);
      }
      __syncthreads();   // old S slice fully read before overwrite

      // Phase B: S[i0..i0+127][:] = G_T (.) S0 + k^T (128x32) @ V (32x64).
#pragma unroll
      for (int q4 = 0; q4 < 4; ++q4) {
        const int tt  = wave + 8 * q4;
        const int mi  = tt & 7;
        const int nt2 = tt >> 3;
        const int mA    = lane & 15;
        const int koffA = (lane >> 4) << 4;
        v16h af;
#pragma unroll
        for (int v = 0; v < 4; ++v) {
          int k = koffA + 2 * v;
          af[2*v]     = (_Float16)KBsm[(size_t)k * TI_ + mi * 16 + mA];
          af[2*v + 1] = (_Float16)KBsm[(size_t)(k + 1) * TI_ + mi * 16 + mA];
          int k2 = 8 + koffA + 2 * v;
          af[8 + 2*v]     = (_Float16)KBsm[(size_t)k2 * TI_ + mi * 16 + mA];
          af[8 + 2*v + 1] = (_Float16)KBsm[(size_t)(k2 + 1) * TI_ + mi * 16 + mA];
        }
        v16h bf = load_b_frag_f32(Vcur + nt2 * 16, DJ_, lane);

        const int nC    = lane & 15;
        const int rbase = (lane >> 4) << 3;
        v8f cc;
#pragma unroll
        for (int r = 0; r < 8; ++r) {
          const int il = mi * 16 + r + rbase;
          cc[r] = Gsm[(T_ - 1) * TI_ + il] *
                  Ssm[(size_t)(i0 + il) * DJ_ + nt2 * 16 + nC];
        }
        v8f dd = __builtin_amdgcn_wmma_f32_16x16x32_f16(false, af, false, bf,
                                                        (short)0, cc, false, false);
#pragma unroll
        for (int r = 0; r < 8; ++r) {
          const int il = mi * 16 + r + rbase;
          Ssm[(size_t)(i0 + il) * DJ_ + nt2 * 16 + nC] = dd[r];
        }
      }
      __syncthreads();
    } // i-tiles

    // materialize masked P
#pragma unroll
    for (int pp = 0; pp < 4; ++pp) {
      const int pr = tid + 256 * pp;
      const int t = pr >> 5, tau = pr & 31;
      Psm[t * T_ + tau] = (tau <= t) ? pacc[pp] : 0.0f;
    }
    __syncthreads();

    // finalize: num = nacc + tril(P) @ V, out = num / (den + eps)
    {
      const int nC    = lane & 15;
      const int rbase = (lane >> 4) << 3;
      const int j = nt * 16 + nC;
#pragma unroll
      for (int r = 0; r < 8; ++r) {
        const int t = mt * 16 + r + rbase;
        float s = nacc[r];
        const float* Pr = Psm + t * T_;
        for (int tau = 0; tau <= t; ++tau)
          s += Pr[tau] * Vcur[tau * DJ_ + j];
        const float den = DENsm[t] + EPS_;
        out[base + (size_t)(t0 + t) * D_ + j0 + j] = s / den;
      }
    }
    __syncthreads();
  } // chunks
}

// ---------------- host side -------------------------------------------------
extern "C" void kernel_launch(void* const* d_in, const int* in_sizes, int n_in,
                              void* d_out, int out_size, void* d_ws, size_t ws_size,
                              hipStream_t stream) {
  (void)in_sizes; (void)n_in; (void)out_size; (void)ws_size;
  const float* x  = (const float*)d_in[0];
  const float* Wq = (const float*)d_in[1];
  const float* bq = (const float*)d_in[2];
  const float* Wk = (const float*)d_in[3];
  const float* bk = (const float*)d_in[4];
  const float* Wv = (const float*)d_in[5];
  const float* bv = (const float*)d_in[6];
  const float* Wa = (const float*)d_in[7];
  const float* ba = (const float*)d_in[8];
  float* outp = (float*)d_out;

  const size_t N = (size_t)B_ * S_ * D_;
  float* qw = (float*)d_ws;
  float* kw = qw + N;
  float* vw = kw + N;
  float* aw = vw + N;

  dim3 g1((B_ * S_) / 16, (4 * D_) / 128);
  suprag_proj_kernel<<<g1, dim3(256), 0, stream>>>(
      x, Wq, bq, Wk, bk, Wv, bv, Wa, ba, qw, kw, vw, aw);

  const size_t smem = sizeof(float) *
      ((size_t)D_ * DJ_        // S
       + 2u * T_ * DJ_         // V double buffer
       + 6u * T_ * TI_         // q/k/a double buffers
       + 4u * T_ * TI_         // G, QG, KD, KB
       + D_ + T_ * T_ + T_);   // z, P, den
  hipFuncSetAttribute((const void*)suprag_scan_kernel,
                      hipFuncAttributeMaxDynamicSharedMemorySize, (int)smem);
  suprag_scan_kernel<<<dim3(D_ / DJ_, B_), dim3(256), smem, stream>>>(
      qw, kw, vw, aw, outp);
}
